// BigGNN_35287451304537
// MI455X (gfx1250) — matrix operand order repaired and checked by reading, not compiled, not run on hardware
//
#include <hip/hip_runtime.h>
#include <math.h>

// ---------------------------------------------------------------------------
// Types for CDNA5 WMMA (gfx1250, wave32)
// ---------------------------------------------------------------------------
typedef __bf16 bf16_t;
typedef bf16_t v16bf __attribute__((ext_vector_type(16)));
typedef bf16_t v8bf  __attribute__((ext_vector_type(8)));
typedef float  v8f   __attribute__((ext_vector_type(8)));

#define HEADS 4
#define DIMD  300
#define HD    1200   // HEADS * DIMD
#define KPAD  320    // 300 padded to multiple of 32 (zero fill)

__device__ __forceinline__ bf16_t f2bf(float f) {
  // fp32 -> bf16 round-to-nearest-even (used only in one-shot convert kernels)
  union { float f; unsigned u; } in; in.f = f;
  unsigned r = in.u + 0x7FFFu + ((in.u >> 16) & 1u);
  union { unsigned short s; bf16_t b; } out;
  out.s = (unsigned short)(r >> 16);
  return out.b;
}

__device__ __forceinline__ unsigned fenc(float f) {
  unsigned u = __float_as_uint(f);
  return (u & 0x80000000u) ? ~u : (u | 0x80000000u);
}
__device__ __forceinline__ float fdec(unsigned u) {
  return __uint_as_float((u & 0x80000000u) ? (u & 0x7FFFFFFFu) : ~u);
}

// ---------------------------------------------------------------------------
// One-shot converters (fp32 -> bf16), zero-padding K to KPAD.
// ---------------------------------------------------------------------------
// A [M,K] fp32 row-major -> Ab [M,KPAD] bf16 row-major
__global__ void cvt_a_kernel(const float* __restrict__ A, bf16_t* __restrict__ Ab,
                             int M, int K) {
  int idx = blockIdx.x * blockDim.x + threadIdx.x;
  if (idx >= M * KPAD) return;
  int m = idx / KPAD, k = idx - m * KPAD;
  float v = (k < K) ? A[(size_t)m * K + k] : 0.0f;
  Ab[idx] = f2bf(v);
}

// B [K,N] fp32 row-major -> Bt [N,KPAD] bf16 (transposed, padded)
__global__ void cvt_bt_kernel(const float* __restrict__ B, bf16_t* __restrict__ Bt,
                              int K, int N) {
  int idx = blockIdx.x * blockDim.x + threadIdx.x;
  if (idx >= N * KPAD) return;
  int n = idx / KPAD, k = idx - n * KPAD;
  float v = (k < K) ? B[(size_t)k * N + n] : 0.0f;
  Bt[idx] = f2bf(v);
}

// ---------------------------------------------------------------------------
// GEMM: C[M,N] = A[M,K] * B[K,N] + bias[N], bf16 inputs / f32 accumulate.
// A pre-converted to [M,KPAD] bf16, B pre-converted to B^T [N,KPAD] bf16.
// One wave computes a 16x64 tile (4 accumulators): per K-step the loop is
// 2 A-loads + 8 B-loads (all global_load_b128) + 4 WMMAs, branchless
// (row/col clamped, K zero-padded), A fragment reused across 4 WMMAs.
// VGPR layouts used here (contiguous K, same order for A and B):
//   A 16x32 : lane r = row M; elems 0..15 -> K=k0+hf*16+e
//   B 32x16 : lane r = col N; elems 0..15 -> K=k0+hf*16+e
//   D 16x16 : elem j -> M=j+8*hf, N=r
// ---------------------------------------------------------------------------
__global__ __launch_bounds__(32)
void gemm_bf16_kernel(const bf16_t* __restrict__ A, const bf16_t* __restrict__ Bt,
                      const float* __restrict__ bias, float* __restrict__ C,
                      int M, int N) {
  const int t  = threadIdx.x;
  const int hf = t >> 4;
  const int r  = t & 15;
  const int col0 = blockIdx.x * 64;
  const int row0 = blockIdx.y * 16;

  int arow = row0 + r; if (arow >= M) arow = M - 1;   // clamp: discard at store
  const bf16_t* Ap = A + (size_t)arow * KPAD;

  const bf16_t* Bp[4];
#pragma unroll
  for (int i = 0; i < 4; ++i) {
    int bc = col0 + 16 * i + r; if (bc >= N) bc = N - 1;
    Bp[i] = Bt + (size_t)bc * KPAD;
  }

  v8f acc[4] = {};
  for (int k0 = 0; k0 < KPAD; k0 += 32) {
    v8bf a0 = *(const v8bf*)(Ap + k0 + hf * 16);
    v8bf a1 = *(const v8bf*)(Ap + k0 + hf * 16 + 8);
    v16bf av = __builtin_shufflevector(a0, a1, 0,1,2,3,4,5,6,7,8,9,10,11,12,13,14,15);
#pragma unroll
    for (int i = 0; i < 4; ++i) {
      v8bf b0 = *(const v8bf*)(Bp[i] + k0 + hf * 16);
      v8bf b1 = *(const v8bf*)(Bp[i] + k0 + hf * 16 + 8);
      v16bf bv = __builtin_shufflevector(b0, b1, 0,1,2,3,4,5,6,7,8,9,10,11,12,13,14,15);
      acc[i] = __builtin_amdgcn_wmma_f32_16x16x32_bf16(
          false, av, false, bv, (short)0, acc[i], false, false);
    }
  }

#pragma unroll
  for (int i = 0; i < 4; ++i) {
    int col = col0 + 16 * i + r;
    if (col < N) {
      float bb = bias[col];
#pragma unroll
      for (int j = 0; j < 8; ++j) {
        int row = row0 + hf * 8 + j;
        if (row < M) C[(size_t)row * N + col] = acc[i][j] + bb;
      }
    }
  }
}

// ---------------------------------------------------------------------------
// Fill kernels
// ---------------------------------------------------------------------------
__global__ void fill_f32_kernel(float* p, int n, float v) {
  int i = blockIdx.x * blockDim.x + threadIdx.x;
  if (i < n) p[i] = v;
}
__global__ void fill_u32_kernel(unsigned* p, int n, unsigned v) {
  int i = blockIdx.x * blockDim.x + threadIdx.x;
  if (i < n) p[i] = v;
}

// ---------------------------------------------------------------------------
// Sparse attention (edge-list) kernels: warp per (edge, head)
// ---------------------------------------------------------------------------
__global__ void edge_logits_kernel(const float* __restrict__ q, const float* __restrict__ k,
                                   const int* __restrict__ src, const int* __restrict__ dst,
                                   int E, float* __restrict__ logits, unsigned* __restrict__ menc) {
  int warp = (blockIdx.x * blockDim.x + threadIdx.x) >> 5;
  int lane = threadIdx.x & 31;
  if (warp >= E * HEADS) return;
  int e = warp >> 2, h = warp & 3;
  int sn = src[e], dn = dst[e];
  const float* qp = q + (size_t)dn * HD + h * DIMD;
  const float* kp = k + (size_t)sn * HD + h * DIMD;
  float a = 0.0f;
  for (int d = lane; d < DIMD; d += 32) a += qp[d] * kp[d];
#pragma unroll
  for (int off = 16; off > 0; off >>= 1) a += __shfl_xor(a, off, 32);
  if (lane == 0) {
    float lg = a * 0.05773502691896258f;  // 1/sqrt(300)
    logits[warp] = lg;
    atomicMax(&menc[dn * HEADS + h], fenc(lg));
  }
}

__global__ void edge_exp_kernel(float* __restrict__ logits, const int* __restrict__ dst,
                                int E, const unsigned* __restrict__ menc,
                                float* __restrict__ s) {
  int idx = blockIdx.x * blockDim.x + threadIdx.x;
  if (idx >= E * HEADS) return;
  int e = idx >> 2, h = idx & 3;
  int dn = dst[e];
  float mv = fdec(menc[dn * HEADS + h]);
  float ev = expf(logits[idx] - mv);
  logits[idx] = ev;
  atomicAdd(&s[dn * HEADS + h], ev);
}

__global__ void edge_scatter_kernel(const float* __restrict__ logits, const float* __restrict__ v,
                                    const int* __restrict__ src, const int* __restrict__ dst,
                                    int E, const float* __restrict__ s, float* __restrict__ oacc) {
  int warp = (blockIdx.x * blockDim.x + threadIdx.x) >> 5;
  int lane = threadIdx.x & 31;
  if (warp >= E * HEADS) return;
  int e = warp >> 2, h = warp & 3;
  int sn = src[e], dn = dst[e];
  float alpha = logits[warp] / s[dn * HEADS + h];
  const float* vp = v + (size_t)sn * HD + h * DIMD;
  float* op = oacc + (size_t)dn * HD + h * DIMD;
  for (int d = lane; d < DIMD; d += 32) atomicAdd(&op[d], alpha * vp[d]);
}

// ---------------------------------------------------------------------------
// Dense bipartite attention (cross edges are all-pairs): block per (dst, head)
// ---------------------------------------------------------------------------
__global__ __launch_bounds__(256)
void dense_attn_kernel(const float* __restrict__ q, const float* __restrict__ k,
                       const float* __restrict__ v, float* __restrict__ oacc,
                       int n_dst, int n_src, int src_off) {
  __shared__ float sc[320];
  __shared__ float red[256];
  int i = blockIdx.x >> 2;
  int h = blockIdx.x & 3;
  int t = threadIdx.x;
  const float* qp = q + (size_t)i * HD + h * DIMD;

  for (int j = t; j < n_src; j += 256) {
    const float* kp = k + (size_t)(src_off + j) * HD + h * DIMD;
    float a = 0.0f;
    for (int d = 0; d < DIMD; ++d) a += qp[d] * kp[d];
    sc[j] = a * 0.05773502691896258f;
  }
  __syncthreads();

  float mx = -3.0e38f;
  for (int j = t; j < n_src; j += 256) mx = fmaxf(mx, sc[j]);
  red[t] = mx; __syncthreads();
  for (int s2 = 128; s2 > 0; s2 >>= 1) { if (t < s2) red[t] = fmaxf(red[t], red[t + s2]); __syncthreads(); }
  mx = red[0]; __syncthreads();

  float sm = 0.0f;
  for (int j = t; j < n_src; j += 256) { float e = expf(sc[j] - mx); sc[j] = e; sm += e; }
  red[t] = sm; __syncthreads();
  for (int s2 = 128; s2 > 0; s2 >>= 1) { if (t < s2) red[t] += red[t + s2]; __syncthreads(); }
  float inv_s = 1.0f / red[0];
  __syncthreads();

  for (int d = t; d < DIMD; d += 256) {
    float a = 0.0f;
    for (int j = 0; j < n_src; ++j)
      a += sc[j] * v[(size_t)(src_off + j) * HD + h * DIMD + d];
    oacc[(size_t)i * HD + h * DIMD + d] = a * inv_s;
  }
}

// ---------------------------------------------------------------------------
// mean over heads + skip
// ---------------------------------------------------------------------------
__global__ void finalize_kernel(const float* __restrict__ oacc, const float* __restrict__ skip,
                                int n, float* __restrict__ xout) {
  int idx = blockIdx.x * blockDim.x + threadIdx.x;
  if (idx >= n * DIMD) return;
  int node = idx / DIMD, d = idx - node * DIMD;
  const float* oa = oacc + (size_t)node * HD + d;
  float mv = 0.25f * (oa[0] + oa[DIMD] + oa[2 * DIMD] + oa[3 * DIMD]);
  xout[idx] = mv + skip[idx];
}

__global__ __launch_bounds__(256)
void l2norm_kernel(float* __restrict__ x, int nrows) {
  __shared__ float red[256];
  int r = blockIdx.x, t = threadIdx.x;
  if (r >= nrows) return;
  float a = 0.0f;
  for (int d = t; d < DIMD; d += 256) { float v = x[(size_t)r * DIMD + d]; a += v * v; }
  red[t] = a; __syncthreads();
  for (int s2 = 128; s2 > 0; s2 >>= 1) { if (t < s2) red[t] += red[t + s2]; __syncthreads(); }
  float inv = 1.0f / fmaxf(sqrtf(red[0]), 1e-12f);
  for (int d = t; d < DIMD; d += 256) x[(size_t)r * DIMD + d] *= inv;
}

__global__ void mean_pool_kernel(const float* __restrict__ x, int nrows, float* __restrict__ out) {
  int d = blockIdx.x * blockDim.x + threadIdx.x;
  if (d >= DIMD) return;
  float a = 0.0f;
  for (int n = 0; n < nrows; ++n) a += x[(size_t)n * DIMD + d];
  out[d] = a / (float)nrows;
}

// ---------------------------------------------------------------------------
// MLP head: 600 -> 600 -> 300 -> 300 -> 1 (relu, relu, relu, sigmoid)
// ---------------------------------------------------------------------------
__global__ __launch_bounds__(256)
void mlp_kernel(const float* __restrict__ hin,
                const float* __restrict__ W1, const float* __restrict__ b1,
                const float* __restrict__ W2, const float* __restrict__ b2,
                const float* __restrict__ W3, const float* __restrict__ b3,
                const float* __restrict__ W4, const float* __restrict__ b4,
                float* __restrict__ out1) {
  __shared__ float h0[600];
  __shared__ float h1[600];
  __shared__ float red[256];
  int t = threadIdx.x;
  for (int j = t; j < 600; j += 256) h0[j] = hin[j];
  __syncthreads();
  for (int j = t; j < 600; j += 256) {
    float a = b1[j];
    for (int i = 0; i < 600; ++i) a += h0[i] * W1[i * 600 + j];
    h1[j] = fmaxf(a, 0.0f);
  }
  __syncthreads();
  for (int j = t; j < 300; j += 256) {
    float a = b2[j];
    for (int i = 0; i < 600; ++i) a += h1[i] * W2[i * 300 + j];
    h0[j] = fmaxf(a, 0.0f);
  }
  __syncthreads();
  for (int j = t; j < 300; j += 256) {
    float a = b3[j];
    for (int i = 0; i < 300; ++i) a += h0[i] * W3[i * 300 + j];
    h1[j] = fmaxf(a, 0.0f);
  }
  __syncthreads();
  float p = 0.0f;
  for (int i = t; i < 300; i += 256) p += h1[i] * W4[i];
  red[t] = p; __syncthreads();
  for (int s2 = 128; s2 > 0; s2 >>= 1) { if (t < s2) red[t] += red[t + s2]; __syncthreads(); }
  if (t == 0) out1[0] = 1.0f / (1.0f + expf(-(red[0] + b4[0])));
}

// ---------------------------------------------------------------------------
// Host orchestration
// ---------------------------------------------------------------------------
extern "C" void kernel_launch(void* const* d_in, const int* in_sizes, int n_in,
                              void* d_out, int out_size, void* d_ws, size_t ws_size,
                              hipStream_t stream) {
  (void)in_sizes; (void)n_in; (void)out_size; (void)ws_size;
  const int N1 = 300, N2 = 200, E1 = 4800, E2 = 3200;

  const float* x_1  = (const float*)d_in[0];
  const float* x_2  = (const float*)d_in[1];
  const int*   ei1  = (const int*)d_in[2];
  const int*   ei2  = (const int*)d_in[3];
  // d_in[4], d_in[5] (edge_attr) unused by the reference
  const float* Wq   = (const float*)d_in[6];
  const float* bq   = (const float*)d_in[7];
  const float* Wk   = (const float*)d_in[8];
  const float* bk   = (const float*)d_in[9];
  const float* Wv   = (const float*)d_in[10];
  const float* bvp  = (const float*)d_in[11];
  const float* Wsk  = (const float*)d_in[12];
  const float* bsk  = (const float*)d_in[13];
  const float* mW1  = (const float*)d_in[14];
  const float* mb1  = (const float*)d_in[15];
  const float* mW2  = (const float*)d_in[16];
  const float* mb2  = (const float*)d_in[17];
  const float* mW3  = (const float*)d_in[18];
  const float* mb3  = (const float*)d_in[19];
  const float* mW4  = (const float*)d_in[20];
  const float* mb4  = (const float*)d_in[21];
  float* out = (float*)d_out;

  const int* s1 = ei1;        const int* d1 = ei1 + E1;
  const int* s2 = ei2;        const int* d2 = ei2 + E2;

  // Workspace carving
  float* ws = (float*)d_ws;
  size_t off = 0;
  auto alloc = [&](size_t nfloats) {
    float* p = ws + off; off += (nfloats + 63) & ~(size_t)63; return p;
  };
  auto alloc_bf = [&](size_t nbf) { return (bf16_t*)alloc((nbf + 1) / 2); };

  float* x1b   = alloc((size_t)N1 * DIMD);
  float* x2b   = alloc((size_t)N2 * DIMD);
  float* y1    = alloc((size_t)N1 * DIMD);
  float* y2    = alloc((size_t)N2 * DIMD);
  float* xc1   = alloc((size_t)(N1 + N2) * DIMD);
  float* xc2   = alloc((size_t)(N1 + N2) * DIMD);
  float* qb    = alloc((size_t)(N1 + N2) * HD);
  float* kb    = alloc((size_t)(N1 + N2) * HD);
  float* vb    = alloc((size_t)(N1 + N2) * HD);
  float* skipb = alloc((size_t)(N1 + N2) * DIMD);
  float* oacc  = alloc((size_t)(N1 + N2) * HD);
  unsigned* menc = (unsigned*)alloc(2048);
  float* sbuf  = alloc(2048);
  float* lgts  = alloc((size_t)E1 * HEADS);
  // bf16 buffers
  bf16_t* Abf  = alloc_bf((size_t)(N1 + N2) * KPAD);              // activations
  const size_t WBIG = (size_t)HD * KPAD;    // 1200*320
  const size_t WSML = (size_t)DIMD * KPAD;  // 300*320
  bf16_t* WqT = alloc_bf(8 * WBIG);
  bf16_t* WkT = alloc_bf(8 * WBIG);
  bf16_t* WvT = alloc_bf(8 * WBIG);
  bf16_t* WsT = alloc_bf(8 * WSML);

  // --- One-shot weight conversion: fp32 [K,N] -> bf16 transposed [N,KPAD] ---
  {
    int gb = ((int)WBIG + 255) / 256;
    int gs = ((int)WSML + 255) / 256;
    for (int ij = 0; ij < 8; ++ij) {
      cvt_bt_kernel<<<gb, 256, 0, stream>>>(Wq  + (size_t)ij * DIMD * HD,   WqT + (size_t)ij * WBIG, DIMD, HD);
      cvt_bt_kernel<<<gb, 256, 0, stream>>>(Wk  + (size_t)ij * DIMD * HD,   WkT + (size_t)ij * WBIG, DIMD, HD);
      cvt_bt_kernel<<<gb, 256, 0, stream>>>(Wv  + (size_t)ij * DIMD * HD,   WvT + (size_t)ij * WBIG, DIMD, HD);
      cvt_bt_kernel<<<gs, 256, 0, stream>>>(Wsk + (size_t)ij * DIMD * DIMD, WsT + (size_t)ij * WSML, DIMD, DIMD);
    }
  }

  auto gemm = [&](const bf16_t* Ab, const bf16_t* Bt, const float* bias, float* C,
                  int M, int N) {
    dim3 g((N + 63) / 64, (M + 15) / 16);
    gemm_bf16_kernel<<<g, 32, 0, stream>>>(Ab, Bt, bias, C, M, N);
  };

  auto proj_all = [&](const float* x, int Nn, int Mskip, int ij) {
    // convert activations once, then 4 projections
    cvt_a_kernel<<<(Nn * KPAD + 255) / 256, 256, 0, stream>>>(x, Abf, Nn, DIMD);
    gemm(Abf, WqT + (size_t)ij * WBIG, bq  + (size_t)ij * HD,   qb,    Nn,    HD);
    gemm(Abf, WkT + (size_t)ij * WBIG, bk  + (size_t)ij * HD,   kb,    Nn,    HD);
    gemm(Abf, WvT + (size_t)ij * WBIG, bvp + (size_t)ij * HD,   vb,    Nn,    HD);
    gemm(Abf, WsT + (size_t)ij * WSML, bsk + (size_t)ij * DIMD, skipb, Mskip, DIMD);
  };

  auto sparse_conv = [&](const float* x, int Nn, const int* srcp, const int* dstp, int E,
                         int ij, float* xout) {
    proj_all(x, Nn, Nn, ij);
    fill_u32_kernel<<<(Nn * HEADS + 255) / 256, 256, 0, stream>>>(menc, Nn * HEADS, 0u);
    fill_f32_kernel<<<(Nn * HEADS + 255) / 256, 256, 0, stream>>>(sbuf, Nn * HEADS, 0.0f);
    fill_f32_kernel<<<(Nn * HD + 255) / 256, 256, 0, stream>>>(oacc, Nn * HD, 0.0f);
    int warps = E * HEADS;
    edge_logits_kernel<<<(warps + 7) / 8, 256, 0, stream>>>(qb, kb, srcp, dstp, E, lgts, menc);
    edge_exp_kernel<<<(warps + 255) / 256, 256, 0, stream>>>(lgts, dstp, E, menc, sbuf);
    edge_scatter_kernel<<<(warps + 7) / 8, 256, 0, stream>>>(lgts, vb, srcp, dstp, E, sbuf, oacc);
    finalize_kernel<<<(Nn * DIMD + 255) / 256, 256, 0, stream>>>(oacc, skipb, Nn, xout);
  };

  auto cross_conv = [&](const float* xc, int n_dst, int n_src, int ij, float* xout) {
    int Nn = n_dst + n_src;
    proj_all(xc, Nn, n_dst, ij);
    dense_attn_kernel<<<n_dst * HEADS, 256, 0, stream>>>(qb, kb, vb, oacc, n_dst, n_src, n_dst);
    finalize_kernel<<<(n_dst * DIMD + 255) / 256, 256, 0, stream>>>(oacc, skipb, n_dst, xout);
  };

  // Initialize activations
  hipMemcpyAsync(x1b, x_1, (size_t)N1 * DIMD * sizeof(float), hipMemcpyDeviceToDevice, stream);
  hipMemcpyAsync(x2b, x_2, (size_t)N2 * DIMD * sizeof(float), hipMemcpyDeviceToDevice, stream);

  for (int L = 0; L < 2; ++L) {
    // Self-attention convs
    sparse_conv(x1b, N1, s1, d1, E1, L * 4 + 0, y1);
    sparse_conv(x2b, N2, s2, d2, E2, L * 4 + 1, y2);
    // Build concats: xc1 = [y1; y2], xc2 = [y2; y1]
    hipMemcpyAsync(xc1, y1, (size_t)N1 * DIMD * sizeof(float), hipMemcpyDeviceToDevice, stream);
    hipMemcpyAsync(xc1 + (size_t)N1 * DIMD, y2, (size_t)N2 * DIMD * sizeof(float),
                   hipMemcpyDeviceToDevice, stream);
    hipMemcpyAsync(xc2, y2, (size_t)N2 * DIMD * sizeof(float), hipMemcpyDeviceToDevice, stream);
    hipMemcpyAsync(xc2 + (size_t)N2 * DIMD, y1, (size_t)N1 * DIMD * sizeof(float),
                   hipMemcpyDeviceToDevice, stream);
    // Cross-attention convs (dense bipartite)
    cross_conv(xc1, N1, N2, L * 4 + 2, x1b);
    cross_conv(xc2, N2, N1, L * 4 + 3, x2b);
    l2norm_kernel<<<N1, 256, 0, stream>>>(x1b, N1);
    l2norm_kernel<<<N2, 256, 0, stream>>>(x2b, N2);
  }

  // Readout: means into d_out[0..299] and d_out[300..599], then MLP -> d_out[600]
  mean_pool_kernel<<<(DIMD + 255) / 256, 256, 0, stream>>>(x1b, N1, out);
  mean_pool_kernel<<<(DIMD + 255) / 256, 256, 0, stream>>>(x2b, N2, out + DIMD);
  mlp_kernel<<<1, 256, 0, stream>>>(out, mW1, mb1, mW2, mb2, mW3, mb3, mW4, mb4, out + 2 * DIMD);
}
